// LSTMRecActor_69174743269834
// MI455X (gfx1250) — compile-verified, matrix operand order checked by reading, not run on hardware
//
#include <hip/hip_runtime.h>
#include <hip/hip_bf16.h>

typedef __attribute__((ext_vector_type(16))) _Float16 v16h;
typedef __attribute__((ext_vector_type(8)))  float    v8f;

#define XT_OFF_BYTES (512*1024)
#define ONES_OFF (16*520)   // one-hot [1,0,...] row appended to xbuf

#if __has_builtin(__builtin_amdgcn_rcpf)
#define FAST_RCP(x) __builtin_amdgcn_rcpf(x)
#else
#define FAST_RCP(x) (1.f/(x))
#endif

#if __has_builtin(__builtin_amdgcn_exp2f)
#define EXP2F(x) __builtin_amdgcn_exp2f(x)
#else
#define EXP2F(x) __expf((x)*0.6931471805599453f)
#endif

// Native CDNA5 v_tanh_f32 (confirmed in round-3 asm), else exp2+rcp fallback.
#if defined(__gfx1250__) && __has_builtin(__builtin_amdgcn_tanhf)
__device__ __forceinline__ float ftanh(float x){ return __builtin_amdgcn_tanhf(x); }
__device__ __forceinline__ float sigm(float x){ return 0.5f*__builtin_amdgcn_tanhf(0.5f*x) + 0.5f; }
#else
__device__ __forceinline__ float ftanh(float x){ return 2.f*FAST_RCP(1.f + EXP2F(-2.885390081777927f*x)) - 1.f; }
__device__ __forceinline__ float sigm(float x){ return FAST_RCP(1.f + EXP2F(-1.4426950408889634f*x)); }
#endif

struct Params {
  const float* s;        // [B,544]
  const float* s_hist;   // [B,2,544]
  const int*   hist_len; // [B]
  const float* osW[3]; const float* osB[3];
  const float* Wih[3]; const float* bih[3];
  const float* Whh[3]; const float* bhh[3];
  const float* tsW[3]; const float* tsB[3];
  const float* WihO; const float* bihO;
  const float* WhhO; const float* bhhO;
  const float* pi1W; const float* pi1B;
  const float* pi2W; const float* pi2B;
};

__device__ __forceinline__ v8f wmma_f16(v16h a, v16h b, v8f c){
  return __builtin_amdgcn_wmma_f32_16x16x32_f16(false, a, false, b, (short)0, c, false, false);
}
__device__ __forceinline__ v16h ldfrag(const float* wsf, int fi, int lane){
  return *(const v16h*)(wsf + (size_t)fi*256 + lane*8);
}
__device__ __forceinline__ v8f splat8(float x){ v8f r; for(int j=0;j<8;j++) r[j]=x; return r; }
__device__ __forceinline__ v16h zero16h(){ v16h r; for(int i=0;i<16;i++) r[i]=(_Float16)0.f; return r; }
// Load one K=32 A-chunk (16x32 f16) from the row-major LDS h buffer (stride 72 halves).
__device__ __forceinline__ v16h lds_chunk(const _Float16* hb, int kc, int lane){
  int m = lane & 15;
  int off = m*72 + kc*32 + ((lane<16)?0:8);  // in halves
  union { uint4 q[2]; v16h v; } u;
  u.q[0] = *(const uint4*)(hb + off);
  u.q[1] = *(const uint4*)(hb + off + 16);
  return u.v;
}

__device__ __forceinline__ float wget(const float* W, int od, int id, int n, int k){
  return (n < od && k < id) ? W[n*id + k] : 0.f;
}

// Pre-swizzle all weights into per-lane-contiguous f16 B fragments (286 fragments x 1KB).
// Inner x-chunk (kc==2) carries the combined LSTM bias in its K=8 row (A supplies 1.0 there).
__global__ __launch_bounds__(32) void prep_frags(Params P, unsigned* wsu){
  int fi = blockIdx.x;
  int lane = threadIdx.x;
  const float* W = nullptr; int od=0, id=0, nt=0, kbase=0;
  const float* b1 = nullptr; const float* b2 = nullptr;  // bias row for kc==2
  if (fi < 180) {
    int t = fi / 60, r = fi % 60;
    if (r < 48) { nt = r/3; int kc = r%3;
      if (kc < 2) { W = P.Whh[t]; od=256; id=64; kbase = kc*32; }
      else        { W = P.Wih[t]; od=256; id=8;  kbase = 0; b1 = P.bih[t]; b2 = P.bhh[t]; }
    } else if (r < 52) { nt = r-48; W = P.osW[t]; od=64; id=32; kbase=0; }
    else { int rr=r-52; nt=rr/2; W=P.tsW[t]; od=64; id=64; kbase=(rr%2)*32; }
  } else {
    int r = fi - 180;
    if (r < 96) { nt=r/6; int kc=r%6;
      if (kc<4){ W=P.WihO; od=256; id=128; kbase=kc*32; }
      else     { W=P.WhhO; od=256; id=64;  kbase=(kc-4)*32; }
    } else if (r < 104) { int rr=r-96; nt=rr/2; W=P.pi1W; od=64; id=64; kbase=(rr%2)*32; }
    else { int rr=r-104; nt=0; W=P.pi2W; od=2; id=64; kbase=rr*32; }
  }
  int n = nt*16 + (lane & 15);
  for (int j=0;j<8;j++){
    int klocal = 2*(j&3) + ((j>=4)?16:0) + ((lane<16)?0:8);
    int k = kbase + klocal;
    float v0 = (b1 && k   == 8) ? (b1[n] + b2[n]) : wget(W, od, id, n, k);
    float v1 = (b1 && k+1 == 8) ? (b1[n] + b2[n]) : wget(W, od, id, n, k+1);
    union { _Float16 h[2]; unsigned u; } p;
    p.h[0] = (_Float16)v0;
    p.h[1] = (_Float16)v1;
    wsu[(size_t)fi*256 + lane*8 + j] = p.u;
  }
}

__device__ __forceinline__ const float* src_for(const Params& P, int slot, int row){
  return (slot==0) ? (P.s_hist + (size_t)row*1088)
       : (slot==1) ? (P.s_hist + (size_t)row*1088 + 544)
       :             (P.s + (size_t)row*544);
}

// Inner pass: per (16-row batch tile, slot). 4 waves: wave w owns hidden units [16w,16w+16).
__global__ __launch_bounds__(128) void inner_lstm(Params P, const float* wsf, float* xt){
  __shared__ __align__(16) _Float16 xbuf[16*520 + 8];  // staged f16 TS inputs + one-hot row
  __shared__ __align__(16) _Float16 hbuf[2][16*72];    // double-buffered h exchange
  __shared__ int cnt_s[16];                            // valid-step counts (atomic)
  __shared__ int hl_s[16];
  const int slot = blockIdx.y;
  const int tp   = 2 - slot;            // param set t
  const int base = blockIdx.x * 16;
  const int tid  = threadIdx.x;
  const int lane = tid & 31, w = tid >> 5;

  if (tid < 16){ cnt_s[tid] = 0; hl_s[tid] = P.hist_len[base + tid]; }
  __syncthreads();

  // Stage TS inputs: NaN->0, zero-source mask, f32->f16; fuse the valid-step count.
  {
    int r = tid >> 3, seg = tid & 7;    // 16 rows x 8 segments of 64 floats (8 steps each)
    int row = base + r;
    bool zs = (slot < 2) && (P.hist_len[row] < 2);
    const float* src = src_for(P, slot, row) + 32 + seg*64;
    unsigned* dst = (unsigned*)(xbuf + r*520 + seg*64);
    int cnt = 0;
    for (int i=0;i<32;i++){
      float v0 = src[2*i], v1 = src[2*i+1];
      bool ok0 = (v0==v0);
      if ((i & 3) == 0) cnt += ok0 ? 1 : 0;   // step-leader element, pre-mask
      v0 = ok0 ? v0 : 0.f;
      v1 = (v1==v1) ? v1 : 0.f;
      if (zs){ v0 = 0.f; v1 = 0.f; }
      union { _Float16 h[2]; unsigned u; } pk;
      pk.h[0]=(_Float16)v0; pk.h[1]=(_Float16)v1;
      dst[i] = pk.u;
    }
    atomicAdd(&cnt_s[r], zs ? 8 : cnt);       // zero-source: all 64 steps valid -> hidx 63
  }
  if (tid < 8) xbuf[ONES_OFF + tid] = (_Float16)((tid==0) ? 1.f : 0.f);  // bias one-hot
  for (int i=tid;i<16*72;i+=128) hbuf[0][i] = (_Float16)0.f;
  __syncthreads();

  const int mrow = lane & 15;
  const int rsel = (lane >= 16) ? 8 : 0;
  const int rowA = base + mrow;
  const int coln = w*16 + (lane & 15);
  const bool zsrc = (slot < 2) && (hl_s[mrow] < 2);
  const float* srcA = src_for(P, slot, rowA);
  int hidx_v[8];
  for (int j=0;j<8;j++) hidx_v[j] = cnt_s[j + rsel] - 1;

  // Resident weight fragments: 4 gates x 3 K-chunks (h0,h1, x+bias)
  v16h Wf[4][3];
  for (int g=0; g<4; g++){
    int nt = w + 4*g;
    for (int kc=0;kc<3;kc++) Wf[g][kc] = ldfrag(wsf, tp*60 + nt*3 + kc, lane);
  }

  // x_OS = relu(s_OS @ osW^T + osB)
  v8f xOS;
  {
    v16h aOS = zero16h();
    int off0 = (lane<16)?0:8;
    for (int e=0;e<8;e++){
      float v0 = srcA[off0+e];
      float v1 = srcA[off0+16+e];
      aOS[e]   = (_Float16)(zsrc ? 0.f : v0);
      aOS[8+e] = (_Float16)(zsrc ? 0.f : v1);
    }
    v8f c = splat8(P.osB[tp][coln]);
    c = wmma_f16(aOS, ldfrag(wsf, tp*60 + 48 + w, lane), c);
    for (int j=0;j<8;j++) xOS[j] = fmaxf(c[j], 0.f);
  }

  // 64-step LSTM: weights resident in VGPRs, x (and bias one-hot) from staged LDS,
  // h double-buffered in LDS, first WMMA per gate uses inline-0 C (bias folded into K=8).
  const int xbase = (lane < 16) ? mrow*520 : ONES_OFF;
  const int xstep = (lane < 16) ? 8 : 0;
  const v8f vzero = splat8(0.f);
  v8f cst = splat8(0.f), hcap = splat8(0.f);
  v16h ax = zero16h();                  // elements 8..15 stay zero forever
  for (int t=0;t<64;t++){
    {
      union { uint4 q; _Float16 h[8]; } u;
      u.q = *(const uint4*)(xbuf + xbase + t*xstep);
      for (int e=0;e<8;e++) ax[e] = u.h[e];
    }
    const _Float16* hb = hbuf[t & 1];
    _Float16*       hw = hbuf[(t & 1) ^ 1];
    v16h ah0 = lds_chunk(hb, 0, lane);
    v16h ah1 = lds_chunk(hb, 1, lane);
    v8f acc[4];
    for (int g=0; g<4; g++){
      v8f a = wmma_f16(ax,  Wf[g][2], vzero);   // x + bias (inline-0 C)
      a = wmma_f16(ah0, Wf[g][0], a);
      a = wmma_f16(ah1, Wf[g][1], a);
      acc[g] = a;
    }
    for (int j=0;j<8;j++){
      float ig = sigm(acc[0][j]);
      float fg = sigm(acc[1][j]);
      float gg = ftanh(acc[2][j]);
      float og = sigm(acc[3][j]);
      float cc = fg*cst[j] + ig*gg;
      cst[j] = cc;
      float h = og * ftanh(cc);
      hcap[j] = (hidx_v[j]==t) ? h : hcap[j];
      hw[(j+rsel)*72 + coln] = (_Float16)h;
    }
    __syncthreads();
  }

  // x_TS = relu(h_sel @ tsW^T + tsB)
  for (int j=0;j<8;j++) hbuf[0][(j+rsel)*72 + coln] = (_Float16)hcap[j];
  __syncthreads();
  v8f xTS;
  {
    v16h a0 = lds_chunk(hbuf[0], 0, lane);
    v16h a1 = lds_chunk(hbuf[0], 1, lane);
    v8f c = splat8(P.tsB[tp][coln]);
    c = wmma_f16(a0, ldfrag(wsf, tp*60 + 52 + w*2 + 0, lane), c);
    c = wmma_f16(a1, ldfrag(wsf, tp*60 + 52 + w*2 + 1, lane), c);
    for (int j=0;j<8;j++) xTS[j] = fmaxf(c[j], 0.f);
  }
  for (int j=0;j<8;j++){
    int m = j + rsel;
    float* dst = xt + ((size_t)(base+m)*3 + slot)*128;
    dst[coln]      = xOS[j];
    dst[64 + coln] = xTS[j];
  }
}

// Outer pass: 3-step LSTM (in=128, hid=64) with per-row hist rotation at load, then heads.
__global__ __launch_bounds__(128) void outer_lstm(Params P, const float* wsf, const float* xt, float* out){
  __shared__ __align__(16) _Float16 hbuf[2][16*72];
  __shared__ int hl_s[16];
  const int base = blockIdx.x*16;
  const int tid = threadIdx.x, lane = tid&31, w = tid>>5;
  if (tid < 16) hl_s[tid] = P.hist_len[base+tid];
  for (int i=tid;i<16*72;i+=128) hbuf[0][i] = (_Float16)0.f;
  __syncthreads();
  const int mrow = lane&15, rsel = (lane>=16)?8:0;
  const int coln = w*16 + (lane&15);
  const int hlA = hl_s[mrow];
  int hidx_v[8];
  for (int j=0;j<8;j++) hidx_v[j] = hl_s[j+rsel];
  v8f bias_v[4];
  for (int g=0; g<4; g++){
    int col = g*64 + coln;
    bias_v[g] = splat8(P.bihO[col] + P.bhhO[col]);
  }
  v8f cst = splat8(0.f), hcap = splat8(0.f);
  for (int t=0;t<3;t++){
    int slot = (t + 2 - hlA) % 3;       // idx[b,t] = (t - hist_len + 2) mod 3, nonneg
    const float* xr = xt + ((size_t)(base+mrow)*3 + slot)*128;
    v16h axc[4];
    for (int c4=0;c4<4;c4++){
      int off0 = c4*32 + ((lane<16)?0:8);
      for (int e=0;e<8;e++){
        axc[c4][e]   = (_Float16)xr[off0+e];
        axc[c4][8+e] = (_Float16)xr[off0+16+e];
      }
    }
    const _Float16* hb = hbuf[t & 1];
    _Float16*       hw = hbuf[(t & 1) ^ 1];
    v16h ah0 = lds_chunk(hb,0,lane);
    v16h ah1 = lds_chunk(hb,1,lane);
    v8f acc[4];
    for (int g=0;g<4;g++){
      int fib = 180 + (w + 4*g)*6;
      v8f a = wmma_f16(axc[0], ldfrag(wsf, fib+0, lane), bias_v[g]);
      for (int c4=1;c4<4;c4++) a = wmma_f16(axc[c4], ldfrag(wsf, fib+c4, lane), a);
      a = wmma_f16(ah0, ldfrag(wsf, fib+4, lane), a);
      a = wmma_f16(ah1, ldfrag(wsf, fib+5, lane), a);
      acc[g] = a;
    }
    for (int j=0;j<8;j++){
      float ig = sigm(acc[0][j]);
      float fg = sigm(acc[1][j]);
      float gg = ftanh(acc[2][j]);
      float og = sigm(acc[3][j]);
      float cc = fg*cst[j] + ig*gg; cst[j]=cc;
      float h = og*ftanh(cc);
      hcap[j] = (hidx_v[j]==t) ? h : hcap[j];
      hw[(j+rsel)*72 + coln] = (_Float16)h;
    }
    __syncthreads();
  }
  // pi1: relu(hm @ pi1W^T + pi1B)
  for (int j=0;j<8;j++) hbuf[0][(j+rsel)*72 + coln] = (_Float16)hcap[j];
  __syncthreads();
  v8f x1;
  {
    v16h a0 = lds_chunk(hbuf[0],0,lane);
    v16h a1 = lds_chunk(hbuf[0],1,lane);
    v8f c = splat8(P.pi1B[coln]);
    c = wmma_f16(a0, ldfrag(wsf, 276 + w*2 + 0, lane), c);
    c = wmma_f16(a1, ldfrag(wsf, 276 + w*2 + 1, lane), c);
    for (int j=0;j<8;j++) x1[j] = fmaxf(c[j], 0.f);
  }
  for (int j=0;j<8;j++) hbuf[1][(j+rsel)*72 + coln] = (_Float16)x1[j];
  __syncthreads();
  // pi2: tanh(x1 @ pi2W^T + pi2B), only 2 output cols; wave 0 does it.
  if (w == 0){
    v16h a0 = lds_chunk(hbuf[1],0,lane);
    v16h a1 = lds_chunk(hbuf[1],1,lane);
    int cp = lane & 15;
    v8f c = splat8((cp<2) ? P.pi2B[cp] : 0.f);
    c = wmma_f16(a0, ldfrag(wsf, 284, lane), c);
    c = wmma_f16(a1, ldfrag(wsf, 285, lane), c);
    if (cp < 2){
      for (int j=0;j<8;j++){
        int m = j + rsel;
        out[(size_t)(base+m)*2 + cp] = ftanh(c[j]);
      }
    }
  }
}

extern "C" void kernel_launch(void* const* d_in, const int* in_sizes, int n_in,
                              void* d_out, int out_size, void* d_ws, size_t ws_size,
                              hipStream_t stream) {
  Params P;
  P.s        = (const float*)d_in[0];
  P.s_hist   = (const float*)d_in[1];
  P.hist_len = (const int*)  d_in[3];
  for (int t=0;t<3;t++){
    P.osW[t] = (const float*)d_in[4+t*8+0]; P.osB[t] = (const float*)d_in[4+t*8+1];
    P.Wih[t] = (const float*)d_in[4+t*8+2]; P.bih[t] = (const float*)d_in[4+t*8+3];
    P.Whh[t] = (const float*)d_in[4+t*8+4]; P.bhh[t] = (const float*)d_in[4+t*8+5];
    P.tsW[t] = (const float*)d_in[4+t*8+6]; P.tsB[t] = (const float*)d_in[4+t*8+7];
  }
  P.WihO = (const float*)d_in[28]; P.bihO = (const float*)d_in[29];
  P.WhhO = (const float*)d_in[30]; P.bhhO = (const float*)d_in[31];
  P.pi1W = (const float*)d_in[32]; P.pi1B = (const float*)d_in[33];
  P.pi2W = (const float*)d_in[34]; P.pi2B = (const float*)d_in[35];

  const int B = in_sizes[3];
  float*    wsf = (float*)d_ws;
  unsigned* wsu = (unsigned*)d_ws;
  float*    xt  = (float*)((char*)d_ws + XT_OFF_BYTES);  // [B,3,128] f32 staging

  prep_frags<<<dim3(286), 32, 0, stream>>>(P, wsu);
  inner_lstm<<<dim3(B/16, 3), 128, 0, stream>>>(P, wsf, xt);
  outer_lstm<<<dim3(B/16), 128, 0, stream>>>(P, wsf, xt, (float*)d_out);
}
